// Net_79130477461835
// MI455X (gfx1250) — compile-verified, matrix operand order checked by reading, not run on hardware
//
#include <hip/hip_runtime.h>
#include <hip/hip_bf16.h>

typedef __attribute__((ext_vector_type(16))) _Float16 v16h;
typedef __attribute__((ext_vector_type(8)))  _Float16 v8h;
typedef __attribute__((ext_vector_type(8)))  float    v8f;

#define SPLINE_K 5

// ---------------------------------------------------------------------------
// Spline basis (degree-1 open B-spline, dim=2) computed inline per edge.
// ---------------------------------------------------------------------------
__device__ __forceinline__ void spline_basis(float p0, float p1,
                                             float* bas, int* wi) {
  float v0 = fminf(fmaxf(p0, 0.f), 1.f) * (SPLINE_K - 1);
  float v1 = fminf(fmaxf(p1, 0.f), 1.f) * (SPLINE_K - 1);
  float lo0 = fminf(fmaxf(floorf(v0), 0.f), (float)(SPLINE_K - 2));
  float lo1 = fminf(fmaxf(floorf(v1), 0.f), (float)(SPLINE_K - 2));
  float f0 = v0 - lo0, f1 = v1 - lo1;
  int i0 = (int)lo0, i1 = (int)lo1;
  int t = 0;
  #pragma unroll
  for (int b0 = 0; b0 < 2; ++b0)
    #pragma unroll
    for (int b1 = 0; b1 < 2; ++b1) {
      float w0 = b0 ? f0 : 1.f - f0;
      float w1 = b1 ? f1 : 1.f - f1;
      bas[t] = w0 * w1;
      wi[t]  = i0 + b0 + (i1 + b1) * SPLINE_K;
      ++t;
    }
}

__device__ __forceinline__ float elu(float h) {
  return h > 0.f ? h : (expf(h) - 1.f);
}

// ---------------------------------------------------------------------------
// Convert W (25,IN,OUT=64) + root (IN,OUT) -> f16 B operand pre-swizzled into
// the WMMA per-lane register layout:
//   Wswz[(((kern*nkb + kb)*4 + ot)*32 + lane)*16 + i] = W[kern][k][o]
//   k = kb*32 + (i&7) + (lane>>4)*8 + (i>>3)*16 ,  o = ot*16 + (lane&15)
// so the GEMM reads each lane's 16 halves as two contiguous b128 loads.
// ---------------------------------------------------------------------------
__global__ void conv_w_swz(const float* __restrict__ W,
                           const float* __restrict__ root,
                           _Float16* __restrict__ Wswz, int IN) {
  const int OUT = 64;
  int nkb = IN >> 5;                       // IN/32
  int total = 26 * nkb * 4 * 32 * 16;
  int idx = blockIdx.x * blockDim.x + threadIdx.x;
  if (idx >= total) return;
  int i    = idx & 15;
  int lane = (idx >> 4) & 31;
  int ot   = (idx >> 9) & 3;
  int rest = idx >> 11;                    // kern*nkb + kb
  int kb   = rest % nkb;
  int kern = rest / nkb;
  int k = kb * 32 + (i & 7) + ((lane >> 4) << 3) + ((i >> 3) << 4);
  int o = ot * 16 + (lane & 15);
  float v = (kern < 25) ? W[((size_t)kern * IN + k) * OUT + o]
                        : root[(size_t)k * OUT + o];
  Wswz[idx] = (_Float16)v;
}

// ---------------------------------------------------------------------------
// WMMA GEMM, IN=32: xW[n,kern,o] = sum_i X[n,i]*W[kern,i,o]
// grid: (nodeTiles, 26), block: 128 (4 waves; wave w handles out-tile w)
// X is padded with zero rows to a multiple of 16 -> no bounds checks.
// ---------------------------------------------------------------------------
__global__ void xw_gemm_in32(const _Float16* __restrict__ X,
                             const _Float16* __restrict__ Wswz,
                             float* __restrict__ xW) {
  const int IN = 32, OUT = 64, NK = 26;
  int tile = blockIdx.x, kern = blockIdx.y;
  int wave = threadIdx.x >> 5;             // out-tile 0..3
  int L    = threadIdx.x & 31;
  int lo = L & 15, hi = (L >> 4) & 1;
  int m = tile * 16 + lo;

  union { v16h v; v8h h[2]; } a, b;
  const _Float16* Ab = X + (size_t)m * IN + hi * 8;
  a.h[0] = *(const v8h*)(Ab);
  a.h[1] = *(const v8h*)(Ab + 16);
  const _Float16* Bb = Wswz + ((((size_t)kern * 4 + wave) * 32) + L) * 16;
  b.h[0] = *(const v8h*)(Bb);
  b.h[1] = *(const v8h*)(Bb + 8);

  v8f c = {};
  c = __builtin_amdgcn_wmma_f32_16x16x32_f16(false, a.v, false, b.v,
                                             (short)0, c, false, false);
  int o = wave * 16 + lo;
  #pragma unroll
  for (int r = 0; r < 8; ++r) {
    int mm = tile * 16 + r + hi * 8;
    xW[((size_t)mm * NK + kern) * OUT + o] = c[r];
  }
}

// ---------------------------------------------------------------------------
// WMMA GEMM, IN=64: two chained K=32 WMMAs per 16x16 tile
// ---------------------------------------------------------------------------
__global__ void xw_gemm_in64(const _Float16* __restrict__ X,
                             const _Float16* __restrict__ Wswz,
                             float* __restrict__ xW) {
  const int IN = 64, OUT = 64, NK = 26;
  int tile = blockIdx.x, kern = blockIdx.y;
  int wave = threadIdx.x >> 5;
  int L    = threadIdx.x & 31;
  int lo = L & 15, hi = (L >> 4) & 1;
  int m = tile * 16 + lo;

  v8f c = {};
  #pragma unroll
  for (int kb = 0; kb < 2; ++kb) {
    union { v16h v; v8h h[2]; } a, b;
    const _Float16* Ab = X + (size_t)m * IN + kb * 32 + hi * 8;
    a.h[0] = *(const v8h*)(Ab);
    a.h[1] = *(const v8h*)(Ab + 16);
    const _Float16* Bb = Wswz + ((((size_t)(kern * 2 + kb) * 4 + wave) * 32) + L) * 16;
    b.h[0] = *(const v8h*)(Bb);
    b.h[1] = *(const v8h*)(Bb + 8);
    c = __builtin_amdgcn_wmma_f32_16x16x32_f16(false, a.v, false, b.v,
                                               (short)0, c, false, false);
  }
  int o = wave * 16 + lo;
  #pragma unroll
  for (int r = 0; r < 8; ++r) {
    int mm = tile * 16 + r + hi * 8;
    xW[((size_t)mm * NK + kern) * OUT + o] = c[r];
  }
}

// ---------------------------------------------------------------------------
// Layer 1 edge kernel (in=1): W1 (25x32) cached in LDS
// ---------------------------------------------------------------------------
__global__ void edge_conv1(const float* __restrict__ x,
                           const int* __restrict__ ei,
                           const float* __restrict__ pseudo,
                           const float* __restrict__ W1,
                           float* __restrict__ agg, float* __restrict__ deg,
                           int E) {
  __shared__ float sW[25 * 32];
  for (int i = threadIdx.x; i < 25 * 32; i += blockDim.x) sW[i] = W1[i];
  __syncthreads();
  int e = blockIdx.x * blockDim.x + threadIdx.x;
  if (e >= E) return;
  int src = ei[e], dst = ei[E + e];
  float bas[4]; int wi[4];
  spline_basis(pseudo[2 * e], pseudo[2 * e + 1], bas, wi);
  float xs = x[src];
  atomicAdd(&deg[dst], 1.0f);
  #pragma unroll 4
  for (int o = 0; o < 32; ++o) {
    float msum = 0.f;
    #pragma unroll
    for (int b = 0; b < 4; ++b) msum += bas[b] * sW[wi[b] * 32 + o];
    atomicAdd(&agg[(size_t)dst * 32 + o], xs * msum);
  }
}

// ---------------------------------------------------------------------------
// Layers 2/3 edge kernel: gather 4 xW slices per edge, scatter-add to dst.
// One wave per edge (lane strides over OUT=64).
// ---------------------------------------------------------------------------
__global__ void edge_convN(const float* __restrict__ xW,
                           const int* __restrict__ ei,
                           const float* __restrict__ pseudo,
                           float* __restrict__ agg, float* __restrict__ deg,
                           int E) {
  const int NK = 26, OUT = 64;
  int e = blockIdx.x * blockDim.y + threadIdx.y;
  if (e >= E) return;
  int lane = threadIdx.x;
  int src = ei[e], dst = ei[E + e];
  float bas[4]; int wi[4];
  spline_basis(pseudo[2 * e], pseudo[2 * e + 1], bas, wi);
  const float* base = xW + (size_t)src * NK * OUT;
  #pragma unroll
  for (int oo = 0; oo < 2; ++oo) {
    int o = lane + oo * 32;
    float msum = 0.f;
    #pragma unroll
    for (int b = 0; b < 4; ++b) msum += bas[b] * base[(size_t)wi[b] * OUT + o];
    atomicAdd(&agg[(size_t)dst * OUT + o], msum);
  }
  if (lane == 0) atomicAdd(&deg[dst], 1.0f);
}

// ---------------------------------------------------------------------------
// Finalize layer 1: mean + root*x + b, ELU, pool into clusters
// ---------------------------------------------------------------------------
__global__ void finalize1(const float* __restrict__ agg,
                          const float* __restrict__ deg,
                          const float* __restrict__ x,
                          const float* __restrict__ root1,
                          const float* __restrict__ b1,
                          const int* __restrict__ cluster,
                          float* __restrict__ pooled, float* __restrict__ cnt,
                          int N) {
  const int OUT = 32;
  int idx = blockIdx.x * blockDim.x + threadIdx.x;
  int n = idx / OUT, o = idx % OUT;
  if (n >= N) return;
  float h = agg[(size_t)n * OUT + o] / fmaxf(deg[n], 1.f)
          + x[n] * root1[o] + b1[o];
  h = elu(h);
  int c = cluster[n];
  atomicAdd(&pooled[(size_t)c * OUT + o], h);
  if (o == 0) atomicAdd(&cnt[c], 1.f);
}

// ---------------------------------------------------------------------------
// Finalize layers 2/3: mean + root term (xW slice 25) + b, ELU, pool
// ---------------------------------------------------------------------------
__global__ void finalizeN(const float* __restrict__ agg,
                          const float* __restrict__ deg,
                          const float* __restrict__ xW,
                          const float* __restrict__ bvec,
                          const int* __restrict__ cluster,
                          float* __restrict__ pooled, float* __restrict__ cnt,
                          int N) {
  const int NK = 26, OUT = 64;
  int idx = blockIdx.x * blockDim.x + threadIdx.x;
  int n = idx / OUT, o = idx % OUT;
  if (n >= N) return;
  float h = agg[(size_t)n * OUT + o] / fmaxf(deg[n], 1.f)
          + xW[((size_t)n * NK + 25) * OUT + o] + bvec[o];
  h = elu(h);
  int c = cluster[n];
  atomicAdd(&pooled[(size_t)c * OUT + o], h);
  if (o == 0) atomicAdd(&cnt[c], 1.f);
}

// ---------------------------------------------------------------------------
// Normalize pooled sums -> f16 node features for next layer's WMMA
// ---------------------------------------------------------------------------
__global__ void norm_to_f16(const float* __restrict__ pooled,
                            const float* __restrict__ cnt,
                            _Float16* __restrict__ xh, int N, int OUT) {
  int idx = blockIdx.x * blockDim.x + threadIdx.x;
  int n = idx / OUT;
  if (n >= N) return;
  xh[idx] = (_Float16)(pooled[idx] / fmaxf(cnt[n], 1.f));
}

// ---------------------------------------------------------------------------
// Head: graph mean -> 64x10 FC -> log_softmax.  grid=B, block=64
// ---------------------------------------------------------------------------
__global__ void head_kernel(const float* __restrict__ gsum,
                            const float* __restrict__ gcnt,
                            const float* __restrict__ fc_w,
                            const float* __restrict__ fc_b,
                            float* __restrict__ out) {
  __shared__ float mean[64];
  __shared__ float logits[10];
  int g = blockIdx.x;
  int t = threadIdx.x;
  mean[t] = gsum[g * 64 + t] / fmaxf(gcnt[g], 1.f);
  __syncthreads();
  if (t < 10) {
    float s = fc_b[t];
    #pragma unroll 8
    for (int o = 0; o < 64; ++o) s += mean[o] * fc_w[o * 10 + t];
    logits[t] = s;
  }
  __syncthreads();
  if (t == 0) {
    float mx = logits[0];
    for (int j = 1; j < 10; ++j) mx = fmaxf(mx, logits[j]);
    float se = 0.f;
    for (int j = 0; j < 10; ++j) se += expf(logits[j] - mx);
    float lse = logf(se) + mx;
    for (int j = 0; j < 10; ++j) out[g * 10 + j] = logits[j] - lse;
  }
}

// ---------------------------------------------------------------------------
// Host side
// ---------------------------------------------------------------------------
static inline size_t alignUp(size_t v) { return (v + 255) & ~(size_t)255; }

extern "C" void kernel_launch(void* const* d_in, const int* in_sizes, int n_in,
                              void* d_out, int out_size, void* d_ws, size_t ws_size,
                              hipStream_t stream) {
  const int N1 = 60000, E1 = 960000;
  const int N2 = 15000, E2 = 240000;
  const int N3 = 4000,  E3 = 64000;
  const int B  = 64;
  const int tiles2 = (N2 + 15) / 16, N2p = tiles2 * 16;   // 938, 15008
  const int tiles3 = (N3 + 15) / 16, N3p = tiles3 * 16;   // 250, 4000

  const float* x       = (const float*)d_in[0];
  const float* pseudo1 = (const float*)d_in[1];
  const float* pseudo2 = (const float*)d_in[2];
  const float* pseudo3 = (const float*)d_in[3];
  const float* W1      = (const float*)d_in[4];
  const float* root1   = (const float*)d_in[5];
  const float* b1      = (const float*)d_in[6];
  const float* W2      = (const float*)d_in[7];
  const float* root2   = (const float*)d_in[8];
  const float* b2      = (const float*)d_in[9];
  const float* W3      = (const float*)d_in[10];
  const float* root3   = (const float*)d_in[11];
  const float* b3      = (const float*)d_in[12];
  const float* fc_w    = (const float*)d_in[13];
  const float* fc_b    = (const float*)d_in[14];
  const int*   ei1     = (const int*)d_in[15];
  const int*   ei2     = (const int*)d_in[16];
  const int*   ei3     = (const int*)d_in[17];
  const int*   cluster1= (const int*)d_in[18];
  const int*   cluster2= (const int*)d_in[19];
  const int*   batch3  = (const int*)d_in[20];
  float* out = (float*)d_out;

  // ---- workspace carve-up ----
  char* ws = (char*)d_ws;
  size_t off = 0;
  auto take = [&](size_t bytes) { char* p = ws + off; off = alignUp(off + bytes); return p; };

  float*    agg1    = (float*)   take((size_t)N1 * 32 * 4);
  float*    deg1    = (float*)   take((size_t)N1 * 4);
  float*    pooled1 = (float*)   take((size_t)N2 * 32 * 4);
  float*    cnt1    = (float*)   take((size_t)N2 * 4);
  _Float16* x2h     = (_Float16*)take((size_t)N2p * 32 * 2);
  _Float16* W2swz   = (_Float16*)take((size_t)26 * 1 * 4 * 32 * 16 * 2);
  float*    xW2     = (float*)   take((size_t)N2p * 26 * 64 * 4);
  float*    agg2    = (float*)   take((size_t)N2 * 64 * 4);
  float*    deg2    = (float*)   take((size_t)N2 * 4);
  float*    pooled2 = (float*)   take((size_t)N3 * 64 * 4);
  float*    cnt2    = (float*)   take((size_t)N3 * 4);
  _Float16* x3h     = (_Float16*)take((size_t)N3p * 64 * 2);
  _Float16* W3swz   = (_Float16*)take((size_t)26 * 2 * 4 * 32 * 16 * 2);
  float*    xW3     = (float*)   take((size_t)N3p * 26 * 64 * 4);
  float*    agg3    = (float*)   take((size_t)N3 * 64 * 4);
  float*    deg3    = (float*)   take((size_t)N3 * 4);
  float*    gsum    = (float*)   take((size_t)B * 64 * 4);
  float*    gcnt    = (float*)   take((size_t)B * 4);
  (void)ws_size; (void)n_in; (void)in_sizes; (void)out_size;

  // ---- zero accumulators + padded f16 features (graph-capture-safe) ----
  hipMemsetAsync(agg1,    0, (size_t)N1 * 32 * 4,  stream);
  hipMemsetAsync(deg1,    0, (size_t)N1 * 4,       stream);
  hipMemsetAsync(pooled1, 0, (size_t)N2 * 32 * 4,  stream);
  hipMemsetAsync(cnt1,    0, (size_t)N2 * 4,       stream);
  hipMemsetAsync(x2h,     0, (size_t)N2p * 32 * 2, stream);
  hipMemsetAsync(agg2,    0, (size_t)N2 * 64 * 4,  stream);
  hipMemsetAsync(deg2,    0, (size_t)N2 * 4,       stream);
  hipMemsetAsync(pooled2, 0, (size_t)N3 * 64 * 4,  stream);
  hipMemsetAsync(cnt2,    0, (size_t)N3 * 4,       stream);
  hipMemsetAsync(x3h,     0, (size_t)N3p * 64 * 2, stream);
  hipMemsetAsync(agg3,    0, (size_t)N3 * 64 * 4,  stream);
  hipMemsetAsync(deg3,    0, (size_t)N3 * 4,       stream);
  hipMemsetAsync(gsum,    0, (size_t)B * 64 * 4,   stream);
  hipMemsetAsync(gcnt,    0, (size_t)B * 4,        stream);

  // ---- weight conversions into WMMA-swizzled f16 layout ----
  {
    int tot2 = 26 * 1 * 4 * 32 * 16;
    conv_w_swz<<<(tot2 + 255) / 256, 256, 0, stream>>>(W2, root2, W2swz, 32);
    int tot3 = 26 * 2 * 4 * 32 * 16;
    conv_w_swz<<<(tot3 + 255) / 256, 256, 0, stream>>>(W3, root3, W3swz, 64);
  }

  // ---- layer 1: per-edge conv (in=1), then finalize+pool ----
  edge_conv1<<<(E1 + 255) / 256, 256, 0, stream>>>(x, ei1, pseudo1, W1,
                                                   agg1, deg1, E1);
  finalize1<<<((size_t)N1 * 32 + 255) / 256, 256, 0, stream>>>(
      agg1, deg1, x, root1, b1, cluster1, pooled1, cnt1, N1);
  norm_to_f16<<<((size_t)N2 * 32 + 255) / 256, 256, 0, stream>>>(
      pooled1, cnt1, x2h, N2, 32);

  // ---- layer 2: WMMA einsum (26 kernels incl. root), edges, finalize ----
  {
    dim3 grid(tiles2, 26);
    xw_gemm_in32<<<grid, 128, 0, stream>>>(x2h, W2swz, xW2);
  }
  {
    dim3 blk(32, 8);
    edge_convN<<<(E2 + 7) / 8, blk, 0, stream>>>(xW2, ei2, pseudo2,
                                                 agg2, deg2, E2);
  }
  finalizeN<<<((size_t)N2 * 64 + 255) / 256, 256, 0, stream>>>(
      agg2, deg2, xW2, b2, cluster2, pooled2, cnt2, N2);
  norm_to_f16<<<((size_t)N3 * 64 + 255) / 256, 256, 0, stream>>>(
      pooled2, cnt2, x3h, N3, 64);

  // ---- layer 3: WMMA einsum, edges, finalize+graph pool ----
  {
    dim3 grid(tiles3, 26);
    xw_gemm_in64<<<grid, 128, 0, stream>>>(x3h, W3swz, xW3);
  }
  {
    dim3 blk(32, 8);
    edge_convN<<<(E3 + 7) / 8, blk, 0, stream>>>(xW3, ei3, pseudo3,
                                                 agg3, deg3, E3);
  }
  finalizeN<<<((size_t)N3 * 64 + 255) / 256, 256, 0, stream>>>(
      agg3, deg3, xW3, b3, batch3, gsum, gcnt, N3);

  // ---- head: mean -> FC -> log_softmax ----
  head_kernel<<<B, 64, 0, stream>>>(gsum, gcnt, fc_w, fc_b, out);
}
